// MultiScaleBrige_10668698763675
// MI455X (gfx1250) — matrix operand order repaired and checked by reading, not compiled, not run on hardware
//
#include <hip/hip_runtime.h>

// MultiScaleBridge ECA — MI455X (gfx1250, wave32) implementation.
//
// Math folding: y_ij = r_w @ (ctx^T @ softmax(Q)) + r_b  and  out += o_seg @ y
// collapse to  out += (o_seg @ r_w @ blockdiag(ctx_h^T)) @ softmax(Q) + o_seg@r_b.
// Large GEMMs run on v_wmma_f32_16x16x32_bf16 (f32 accumulate). The dominant
// activation stream is staged global->LDS with GLOBAL_LOAD_ASYNC_TO_LDS_B128
// (double-buffered, ASYNCcnt-tracked) so the copy of K-tile t+1 overlaps the
// WMMA compute of tile t. Softmaxes / tiny weight compositions stay f32 VALU.
// Peak workspace use ~120 MB (pair a13: Q buffer 4*256*25600 floats).

typedef __attribute__((ext_vector_type(16))) __bf16 v16bf;
typedef __attribute__((ext_vector_type(8)))  float  v8f;

union ABu { v16bf v; unsigned int u[8]; };

__device__ __forceinline__ unsigned short f2bf(float f) {
    union { float f; unsigned int u; } c; c.f = f;
    unsigned int u = c.u;
    u += 0x7FFFu + ((u >> 16) & 1u);   // round-to-nearest-even
    return (unsigned short)(u >> 16);
}

// Async copy of 16 bytes per lane: LDS[lds_off] = MEM[base + voff]  (ASYNCcnt)
__device__ __forceinline__ void async_b128(unsigned lds_off, unsigned voff,
                                           unsigned long long base) {
    asm volatile("global_load_async_to_lds_b128 %0, %1, %2"
                 :: "v"(lds_off), "v"(voff), "s"(base) : "memory");
}
__device__ __forceinline__ void wait_async0() {
    asm volatile("s_wait_asynccnt 0x0" ::: "memory");
}

// ---------------------------------------------------------------------------
// Generic WMMA GEMM:  C[n][M,N] = beta*C + W[n][M,K] @ X[n][K,N] + bias[M]
// W row-major with leading dim ldw; per-batch W via wBatch (0 = shared).
// Requires M%64==0, N%64==0, K%32==0 (true for all call sites here).
// Block = 128 threads (4 waves); block tile 64x64; each wave owns 16 rows.
// ---------------------------------------------------------------------------
__global__ __launch_bounds__(128)
void gemm_wmma_bf16(float* __restrict__ C, long cBatch,
                    const float* __restrict__ W, int ldw, long wBatch,
                    const float* __restrict__ X, long xBatch,
                    const float* __restrict__ bias,
                    int M, int N, int K, int accumulate)
{
    __shared__ float          Bf32[2][32][64];  // async landing pads (16 KB)
    __shared__ unsigned short As[64][36];       // [m][k] bf16, padded
    __shared__ unsigned short Bs[32][68];       // [k][n] bf16, padded

    const int tid  = threadIdx.x;
    const int lane = tid & 31;
    const int wave = tid >> 5;
    const int bz   = blockIdx.z;
    const int m0   = blockIdx.y * 64;
    const int n0   = blockIdx.x * 64;

    const float* Wb = W + (long)bz * wBatch;
    const float* Xb = X + (long)bz * xBatch;
    const unsigned long long xbase = (unsigned long long)(size_t)Xb;

    v8f acc[4];
    #pragma unroll
    for (int i = 0; i < 4; ++i) acc[i] = (v8f){0,0,0,0,0,0,0,0};

    const int nTiles = K >> 5;

    // Issue async copy of B tile 0 into buffer 0 (4 x b128 per thread).
    #pragma unroll
    for (int q = tid; q < 512; q += 128) {
        int k = q >> 4, c4 = (q & 15) << 2;
        unsigned lds = (unsigned)(size_t)&Bf32[0][k][c4];
        unsigned off = (unsigned)(((unsigned)k * (unsigned)N + (unsigned)(n0 + c4)) << 2);
        async_b128(lds, off, xbase);
    }

    for (int it = 0; it < nTiles; ++it) {
        const int k0  = it << 5;
        const int cur = it & 1;

        wait_async0();            // my portion of tile `it` has landed in LDS
        __syncthreads();          // everyone's portion landed; tile it-1 fully consumed

        // Prefetch next B tile into the other buffer (overlaps compute below).
        if (it + 1 < nTiles) {
            #pragma unroll
            for (int q = tid; q < 512; q += 128) {
                int k = q >> 4, c4 = (q & 15) << 2;
                unsigned lds = (unsigned)(size_t)&Bf32[1 - cur][k][c4];
                unsigned off = (unsigned)(((unsigned)(k0 + 32 + k) * (unsigned)N +
                                           (unsigned)(n0 + c4)) << 2);
                async_b128(lds, off, xbase);
            }
        }

        // A tile: 64 rows x 32 k (weights, tiny / L2-hot) — synchronous bf16 stage.
        for (int idx = tid; idx < 64*32; idx += 128) {
            int m = idx >> 5, k = idx & 31;
            As[m][k] = f2bf(Wb[(long)(m0 + m) * ldw + k0 + k]);
        }
        // Convert landed f32 B tile -> bf16 once per tile.
        for (int idx = tid; idx < 32*64; idx += 128) {
            int k = idx >> 6, nn = idx & 63;
            Bs[k][nn] = f2bf(Bf32[cur][k][nn]);
        }
        __syncthreads();

        // A fragment: lane<16 -> K 0..7 & 16..23 ; lane>=16 -> K 8..15 & 24..31
        ABu a;
        {
            int m  = wave * 16 + (lane & 15);
            int kb = (lane >> 4) * 8;
            #pragma unroll
            for (int p = 0; p < 8; ++p) {
                int kk = (p < 4) ? (kb + 2*p) : (16 + kb + 2*(p - 4));
                a.u[p] = (unsigned)As[m][kk] | ((unsigned)As[m][kk + 1] << 16);
            }
        }
        #pragma unroll
        for (int nb = 0; nb < 4; ++nb) {
            ABu bf;                           // B fragment: lane&15 = N, K in A's order
            int nbase = nb * 16 + (lane & 15);
            int kbb = (lane >> 4) * 8;
            #pragma unroll
            for (int p = 0; p < 8; ++p) {
                int kk = (p < 4) ? (kbb + 2*p) : (16 + kbb + 2*(p - 4));
                bf.u[p] = (unsigned)Bs[kk][nbase] |
                          ((unsigned)Bs[kk + 1][nbase] << 16);
            }
            acc[nb] = __builtin_amdgcn_wmma_f32_16x16x32_bf16(
                false, a.v, false, bf.v, (short)0, acc[nb], false, false);
        }
    }

    // C/D layout: M = r + 8*(lane>=16), N = lane&15
    float* Cb = C + (long)bz * cBatch;
    const int hi   = lane >> 4;
    const int col0 = n0 + (lane & 15);
    #pragma unroll
    for (int nb = 0; nb < 4; ++nb) {
        int col = col0 + nb * 16;
        #pragma unroll
        for (int r = 0; r < 8; ++r) {
            int m = m0 + wave * 16 + r + 8 * hi;
            float v = acc[nb][r];
            if (bias) v += bias[m];
            float* cp = Cb + (long)m * N + col;
            if (accumulate) v += *cp;
            *cp = v;
        }
    }
}

// ---------------------------------------------------------------------------
// ctx[n,h] = Ksm_h (cjh x hw) @ V_h^T (hw x cjh)   — tall-K WMMA reduction.
// Block = 128 threads (4 waves), one 32x32 output region per block.
// ---------------------------------------------------------------------------
__global__ __launch_bounds__(128)
void ctx_wmma(float* __restrict__ ctx, const float* __restrict__ Ksm,
              const float* __restrict__ V, int cj, int cjh, int hw)
{
    __shared__ unsigned short Ks[32][68];   // [m][s]
    __shared__ unsigned short Vt[64][36];   // [s][v] (transposed stage)

    const int tiles = cjh >> 5;
    const int tm = (blockIdx.x / tiles) * 32;
    const int tn = (blockIdx.x % tiles) * 32;
    const int h  = blockIdx.y;
    const int bz = blockIdx.z;
    const int tid = threadIdx.x, lane = tid & 31, wave = tid >> 5;
    const int mq = (wave >> 1) * 16;
    const int nq = (wave & 1) * 16;

    const float* Kb = Ksm + ((long)bz * cj + (long)h * cjh + tm) * hw;
    const float* Vb = V   + ((long)bz * cj + (long)h * cjh + tn) * hw;

    v8f acc = (v8f){0,0,0,0,0,0,0,0};

    for (int s0 = 0; s0 < hw; s0 += 64) {
        __syncthreads();
        for (int idx = tid; idx < 32*64; idx += 128) {
            int r = idx >> 6, s = idx & 63;            // r = row (m or v)
            Ks[r][s] = f2bf(Kb[(long)r * hw + s0 + s]);
            Vt[s][r] = f2bf(Vb[(long)r * hw + s0 + s]);
        }
        __syncthreads();
        #pragma unroll
        for (int sc = 0; sc < 2; ++sc) {
            ABu a, bf;
            int m  = mq + (lane & 15);
            int kb = (lane >> 4) * 8;
            #pragma unroll
            for (int p = 0; p < 8; ++p) {
                int kk = sc * 32 + ((p < 4) ? (kb + 2*p) : (16 + kb + 2*(p - 4)));
                a.u[p] = (unsigned)Ks[m][kk] | ((unsigned)Ks[m][kk + 1] << 16);
            }
            int sr = nq + (lane & 15);
            #pragma unroll
            for (int p = 0; p < 8; ++p) {
                int kk = sc * 32 + ((p < 4) ? (kb + 2*p) : (16 + kb + 2*(p - 4)));
                bf.u[p] = (unsigned)Vt[kk][sr] |
                          ((unsigned)Vt[kk + 1][sr] << 16);
            }
            acc = __builtin_amdgcn_wmma_f32_16x16x32_bf16(
                false, a.v, false, bf.v, (short)0, acc, false, false);
        }
    }

    long base = (((long)bz * 2 + h) * cjh) * cjh;
    int col = tn + nq + (lane & 15);
    int hi  = lane >> 4;
    #pragma unroll
    for (int r = 0; r < 8; ++r) {
        int m = tm + mq + r + 8 * hi;
        ctx[base + (long)m * cjh + col] = acc[r];
    }
}

// ---------------------------------------------------------------------------
// Softmax over spatial dim for each (batch,channel) row of K. In place.
// ---------------------------------------------------------------------------
__global__ void softmax_rows(float* __restrict__ X, int len)
{
    float* p = X + (long)blockIdx.x * len;
    __shared__ float red[256];
    const int tid = threadIdx.x;

    float mx = -INFINITY;
    for (int i = tid; i < len; i += 256) mx = fmaxf(mx, p[i]);
    red[tid] = mx; __syncthreads();
    for (int s = 128; s > 0; s >>= 1) { if (tid < s) red[tid] = fmaxf(red[tid], red[tid + s]); __syncthreads(); }
    mx = red[0]; __syncthreads();

    float sum = 0.f;
    for (int i = tid; i < len; i += 256) sum += __expf(p[i] - mx);
    red[tid] = sum; __syncthreads();
    for (int s = 128; s > 0; s >>= 1) { if (tid < s) red[tid] += red[tid + s]; __syncthreads(); }
    float inv = 1.0f / red[0];

    for (int i = tid; i < len; i += 256) p[i] = __expf(p[i] - mx) * inv;
}

// ---------------------------------------------------------------------------
// Softmax over per-head channels (cjh) for Q, per (n, head, spatial). In place.
// ---------------------------------------------------------------------------
__global__ void softmax_chan(float* __restrict__ Q, int cjh, int hw, int total)
{
    int t = blockIdx.x * blockDim.x + threadIdx.x;
    if (t >= total) return;
    int s  = t % hw;
    int nh = t / hw;                     // n*2 + h
    float* p = Q + (long)nh * cjh * hw + s;

    float mx = -INFINITY;
    for (int c = 0; c < cjh; ++c) mx = fmaxf(mx, p[(long)c * hw]);
    float sum = 0.f;
    for (int c = 0; c < cjh; ++c) sum += __expf(p[(long)c * hw] - mx);
    float inv = 1.0f / sum;
    for (int c = 0; c < cjh; ++c) p[(long)c * hw] = __expf(p[(long)c * hw] - mx) * inv;
}

// W2[n][c, h*cjh+k] = sum_v r_w[c, h*cjh+v] * ctx[n,h][k,v]
__global__ void compose_w2(float* __restrict__ W2, const float* __restrict__ ctx,
                           const float* __restrict__ rw, int cj, int cjh)
{
    int t = blockIdx.x * blockDim.x + threadIdx.x;
    int total = 4 * cj * cj;
    if (t >= total) return;
    int kg = t % cj;
    int c  = (t / cj) % cj;
    int n  = t / (cj * cj);
    int h = kg / cjh, k = kg % cjh;
    const float* ct = ctx + (((long)n * 2 + h) * cjh + k) * cjh;
    const float* rp = rw + (long)c * cj + h * cjh;
    float s = 0.f;
    for (int v = 0; v < cjh; ++v) s += rp[v] * ct[v];
    W2[t] = s;
}

// W3[n][c,k] = sum_t o_w[c, seg+t] * W2[n][t,k]
__global__ void compose_w3(float* __restrict__ W3, const float* __restrict__ W2,
                           const float* __restrict__ ow, int owld, int seg,
                           int ci, int cj)
{
    int t = blockIdx.x * blockDim.x + threadIdx.x;
    int total = 4 * ci * cj;
    if (t >= total) return;
    int k = t % cj;
    int c = (t / cj) % ci;
    int n = t / (ci * cj);
    const float* op = ow + (long)c * owld + seg;
    const float* wp = W2 + (long)n * cj * cj;
    float s = 0.f;
    for (int tt = 0; tt < cj; ++tt) s += op[tt] * wp[(long)tt * cj + k];
    W3[t] = s;
}

// b3[c] = sum_t o_w[c, seg+t] * r_b[t]
__global__ void compose_b3(float* __restrict__ b3, const float* __restrict__ ow,
                           int owld, int seg, const float* __restrict__ rb,
                           int ci, int cj)
{
    int c = blockIdx.x * blockDim.x + threadIdx.x;
    if (c >= ci) return;
    const float* op = ow + (long)c * owld + seg;
    float s = 0.f;
    for (int t = 0; t < cj; ++t) s += op[t] * rb[t];
    b3[c] = s;
}

// ---------------------------------------------------------------------------
extern "C" void kernel_launch(void* const* d_in, const int* in_sizes, int n_in,
                              void* d_out, int out_size, void* d_ws, size_t ws_size,
                              hipStream_t stream)
{
    (void)in_sizes; (void)n_in; (void)out_size; (void)ws_size;

    const float* X[4]  = {nullptr, (const float*)d_in[0], (const float*)d_in[1], (const float*)d_in[2]};
    const int    C[4]  = {0, 64, 128, 256};
    const int    HW[4] = {0, 160*160, 80*80, 40*40};
    const long   outOff[4] = {0, 0, 4L*64*25600, 4L*64*25600 + 4L*128*6400};
    const float* OW[4] = {nullptr, (const float*)d_in[51], (const float*)d_in[53], (const float*)d_in[55]};
    const float* OB[4] = {nullptr, (const float*)d_in[52], (const float*)d_in[54], (const float*)d_in[56]};
    float* out = (float*)d_out;
    float* ws  = (float*)d_ws;
    const int B = 4, OWLD = 448;

    // out_i = o_i_w[:, 0:ci] @ x_i + o_i_b   (beta=0: initializes all of d_out)
    for (int i = 1; i <= 3; ++i) {
        dim3 g(HW[i] / 64, C[i] / 64, B);
        gemm_wmma_bf16<<<g, 128, 0, stream>>>(out + outOff[i], (long)C[i] * HW[i],
            OW[i], OWLD, 0L, X[i], (long)C[i] * HW[i], OB[i], C[i], HW[i], C[i], 0);
    }

    struct PairCfg { int i, j, pbase, seg; };
    const PairCfg P[6] = {{1,2,3,64},{1,3,11,192},{2,1,19,128},
                          {2,3,27,192},{3,1,35,256},{3,2,43,320}};

    for (int p = 0; p < 6; ++p) {
        const int i = P[p].i, j = P[p].j;
        const int ci = C[i], cj = C[j], hwi = HW[i], hwj = HW[j], cjh = cj / 2;
        const float* kw = (const float*)d_in[P[p].pbase + 0];
        const float* kb = (const float*)d_in[P[p].pbase + 1];
        const float* qw = (const float*)d_in[P[p].pbase + 2];
        const float* qb = (const float*)d_in[P[p].pbase + 3];
        const float* vw = (const float*)d_in[P[p].pbase + 4];
        const float* vb = (const float*)d_in[P[p].pbase + 5];
        const float* rw = (const float*)d_in[P[p].pbase + 6];
        const float* rb = (const float*)d_in[P[p].pbase + 7];

        float* Qb  = ws;
        float* Kbf = Qb  + (size_t)B * cj * hwi;
        float* Vbf = Kbf + (size_t)B * cj * hwj;
        float* Ctx = Vbf + (size_t)B * cj * hwj;
        float* W2  = Ctx + (size_t)B * 2 * cjh * cjh;
        float* W3  = W2  + (size_t)B * cj * cj;
        float* B3  = W3  + (size_t)B * ci * cj;

        // Q = q_w @ x_i + q_b
        { dim3 g(hwi / 64, cj / 64, B);
          gemm_wmma_bf16<<<g, 128, 0, stream>>>(Qb, (long)cj * hwi, qw, ci, 0L,
              X[i], (long)ci * hwi, qb, cj, hwi, ci, 0); }
        // K, V = {k_w, v_w} @ x_j + bias
        { dim3 g(hwj / 64, cj / 64, B);
          gemm_wmma_bf16<<<g, 128, 0, stream>>>(Kbf, (long)cj * hwj, kw, cj, 0L,
              X[j], (long)cj * hwj, kb, cj, hwj, cj, 0);
          gemm_wmma_bf16<<<g, 128, 0, stream>>>(Vbf, (long)cj * hwj, vw, cj, 0L,
              X[j], (long)cj * hwj, vb, cj, hwj, cj, 0); }

        softmax_rows<<<B * cj, 256, 0, stream>>>(Kbf, hwj);
        { int total = B * 2 * hwi;
          softmax_chan<<<(total + 255) / 256, 256, 0, stream>>>(Qb, cjh, hwi, total); }

        { int tiles = cjh / 32; dim3 g(tiles * tiles, 2, B);
          ctx_wmma<<<g, 128, 0, stream>>>(Ctx, Kbf, Vbf, cj, cjh, hwj); }

        { int total = B * cj * cj;
          compose_w2<<<(total + 255) / 256, 256, 0, stream>>>(W2, Ctx, rw, cj, cjh); }
        { int total = B * ci * cj;
          compose_w3<<<(total + 255) / 256, 256, 0, stream>>>(W3, W2, OW[i], OWLD, P[p].seg, ci, cj); }
        compose_b3<<<(ci + 63) / 64, 64, 0, stream>>>(B3, OW[i], OWLD, P[p].seg, rb, ci, cj);

        // out_i += W3[n] @ softmax(Q) + b3
        { dim3 g(hwi / 64, ci / 64, B);
          gemm_wmma_bf16<<<g, 128, 0, stream>>>(out + outOff[i], (long)ci * hwi,
              W3, cj, (long)ci * cj, Qb, (long)cj * hwi, B3, ci, hwi, cj, 1); }
    }
}
